// Attention_54365696033450
// MI455X (gfx1250) — matrix operand order repaired and checked by reading, not compiled, not run on hardware
//
#include <hip/hip_runtime.h>
#include <hip/hip_bf16.h>

typedef __attribute__((ext_vector_type(16))) _Float16 v16h;
typedef __attribute__((ext_vector_type(8)))  _Float16 v8h;
typedef __attribute__((ext_vector_type(8)))  float    v8f;

// ---------------------------------------------------------------------------
// WMMA helpers (gfx1250 wave32, V_WMMA_F32_16X16X32_F16)
// A fragment: lane holds row m = lane%16; halves 0-7  = K[klo..klo+7],
//             halves 8-15 = K[klo+8..klo+15], klo = (lane>=16)*16.
// B fragment: lane holds col n = lane%16; halves 0-15 = K[kb..kb+15],
//             kb = (lane>=16)*16.  (B supplied as N x K row-major = B^T)
// C: VGPR r, lane L -> row (L>=16)*8 + r, col L%16.
// ---------------------------------------------------------------------------
__device__ __forceinline__ v16h cat8(v8h lo, v8h hi) {
  return __builtin_shufflevector(lo, hi, 0,1,2,3,4,5,6,7,8,9,10,11,12,13,14,15);
}

__device__ __forceinline__ v8f wmma_step(v16h a, v16h b, v8f acc) {
  return __builtin_amdgcn_wmma_f32_16x16x32_f16(false, a, false, b,
                                                (short)0, acc, false, false);
}

// Four 16x16 C tiles (M-quad) sharing one B fragment per K step:
// C[t] = A[mrow0 + t*16, :] x B, B given as BT (N x K row-major).
// (No manual double-buffering: the scheduler + 8 waves/block hide latency;
//  manual pipelining only added v_mov copy chains — verified by disasm.)
__device__ __forceinline__ void gemm_tile4(const _Float16* __restrict__ A, int lda,
                                           const _Float16* __restrict__ BT, int ldb,
                                           int mrow0, int nrow, int K, int lane,
                                           v8f acc[4]) {
  const int klo = (lane >> 4) << 4;
  const int kb  = (lane >> 4) << 4;
  const _Float16* ap = A  + (size_t)mrow0 * lda + klo;
  const _Float16* bp = BT + (size_t)nrow  * ldb + kb;
  for (int k = 0; k < K; k += 32) {
    __builtin_prefetch((const void*)(bp + k + 256), 0, 1);  // global_prefetch_b8
    v16h b = *(const v16h*)(bp + k);
    #pragma unroll
    for (int t = 0; t < 4; ++t) {
      const _Float16* at = ap + (size_t)t * 16 * lda;
      v8h alo = *(const v8h*)(at + k);
      v8h ahi = *(const v8h*)(at + k + 8);
      acc[t] = wmma_step(cat8(alo, ahi), b, acc[t]);
    }
  }
}

// ---------------------------------------------------------------------------
// fp32 -> f16 weight conversion
// ---------------------------------------------------------------------------
__global__ void cvt_f32_f16(const float* __restrict__ src,
                            _Float16* __restrict__ dst, int n) {
  int i = blockIdx.x * blockDim.x + threadIdx.x;
  if (i < n) dst[i] = (_Float16)src[i];
}

// ---------------------------------------------------------------------------
// Depthwise 3x3x3 conv (pad 1) + batchnorm, compile-time OD/stride.
// x: (4,128,16,16,16) f32.  Output transposed: outT[b][s][c] f16, s = OD^3.
// ---------------------------------------------------------------------------
template <int OD, int STRIDE>
__global__ void dwconv_bn(const float* __restrict__ x, const float* __restrict__ w,
                          const float* __restrict__ g, const float* __restrict__ bb,
                          const float* __restrict__ m, const float* __restrict__ v,
                          _Float16* __restrict__ outT) {
  constexpr int ns = OD * OD * OD;
  int idx = blockIdx.x * blockDim.x + threadIdx.x;
  if (idx >= 4 * 128 * ns) return;
  int b   = idx / (128 * ns);
  int rem = idx - b * 128 * ns;
  int c   = rem / ns;
  int s   = rem - c * ns;
  int oz  = s / (OD * OD);
  int r2  = s - oz * OD * OD;
  int oy  = r2 / OD;
  int ox  = r2 - oy * OD;

  const float* xc = x + ((size_t)(b * 128 + c)) * 4096;
  const float* wc = w + c * 27;
  float sum = 0.f;
  #pragma unroll
  for (int kz = 0; kz < 3; ++kz) {
    int iz = oz * STRIDE - 1 + kz;
    if (iz < 0 || iz >= 16) continue;
    #pragma unroll
    for (int ky = 0; ky < 3; ++ky) {
      int iy = oy * STRIDE - 1 + ky;
      if (iy < 0 || iy >= 16) continue;
      #pragma unroll
      for (int kx = 0; kx < 3; ++kx) {
        int ix = ox * STRIDE - 1 + kx;
        if (ix < 0 || ix >= 16) continue;
        sum += xc[(iz * 16 + iy) * 16 + ix] * wc[kz * 9 + ky * 3 + kx];
      }
    }
  }
  float bn = (sum - m[c]) * rsqrtf(v[c] + 1e-5f) * g[c] + bb[c];
  outT[((size_t)b * ns + s) * 128 + c] = (_Float16)bn;
}

// ---------------------------------------------------------------------------
// q pointwise conv: q = Wq(512x128) x qdw[b](128x4096), 4 M-tiles per wave.
// Writes qT[b][h][i][d].  grid: 1024 x 256 (8192 wave tasks = 4 x 8 x 256)
// ---------------------------------------------------------------------------
__global__ __launch_bounds__(256) void gemm_q(const _Float16* __restrict__ Wq,
                                              const _Float16* __restrict__ qdwT,
                                              _Float16* __restrict__ qT) {
  int lane = threadIdx.x & 31;
  int task = blockIdx.x * 8 + (threadIdx.x >> 5);
  int b  = task >> 11;
  int t  = task & 2047;
  int mq = t >> 8;             // head h = mq (64 rows per quad)
  int nt = t & 255;
  v8f acc[4] = {};
  gemm_tile4(Wq, 128, qdwT + (size_t)b * 4096 * 128, 128,
             mq * 64 + (lane & 15), nt * 16 + (lane & 15), 128, lane, acc);
  int s  = nt * 16 + (lane & 15);
  int rb = (lane >> 4) << 3;
  #pragma unroll
  for (int tt = 0; tt < 4; ++tt) {
    _Float16* dst = qT + (((size_t)(b * 8 + mq) * 4096) + s) * 64 + tt * 16 + rb;
    #pragma unroll
    for (int r = 0; r < 8; ++r) dst[r] = (_Float16)acc[tt][r];
  }
}

// ---------------------------------------------------------------------------
// kv pointwise conv: kv = Wkv(1024x128) x kvdw[b](128x512), 4 M-tiles/wave.
// mq<8 -> kT[b][h][j][d];  mq>=8 -> vbuf[b][h][d][j]
// grid: 256 x 256 (2048 wave tasks = 4 x 16 x 32)
// ---------------------------------------------------------------------------
__global__ __launch_bounds__(256) void gemm_kv(const _Float16* __restrict__ Wkv,
                                               const _Float16* __restrict__ kvdwT,
                                               _Float16* __restrict__ kT,
                                               _Float16* __restrict__ vbuf) {
  int lane = threadIdx.x & 31;
  int task = blockIdx.x * 8 + (threadIdx.x >> 5);
  int b  = task >> 9;
  int t  = task & 511;
  int mq = t >> 5;             // 0..15
  int nt = t & 31;
  v8f acc[4] = {};
  gemm_tile4(Wkv, 128, kvdwT + (size_t)b * 512 * 128, 128,
             mq * 64 + (lane & 15), nt * 16 + (lane & 15), 128, lane, acc);
  int j  = nt * 16 + (lane & 15);
  int rb = (lane >> 4) << 3;
  if (mq < 8) {                // K rows -> kT[b][h][j][d], d contiguous
    #pragma unroll
    for (int tt = 0; tt < 4; ++tt) {
      _Float16* dst = kT + (((size_t)(b * 8 + mq) * 512) + j) * 64 + tt * 16 + rb;
      #pragma unroll
      for (int r = 0; r < 8; ++r) dst[r] = (_Float16)acc[tt][r];
    }
  } else {                     // V rows -> vbuf[b][h][d][j]
    int h = mq - 8;
    #pragma unroll
    for (int tt = 0; tt < 4; ++tt) {
      #pragma unroll
      for (int r = 0; r < 8; ++r) {
        int d = tt * 16 + rb + r;
        vbuf[(((size_t)(b * 8 + h) * 64) + d) * 512 + j] = (_Float16)acc[tt][r];
      }
    }
  }
}

// ---------------------------------------------------------------------------
// Fused attention: one wave per (b, h, 16-query tile).  LDS = 32 KB: the
// f16 attn matrix is overlaid in place on the f32 dots scratch (per-lane
// forward pack, hazard-free) -> 10 workgroups/WGP.
// attn row i layout: half index i*1024 + chunk*512 + (j & 255), chunk=j>>8.
// ---------------------------------------------------------------------------
__global__ __launch_bounds__(32) void attn_kernel(const _Float16* __restrict__ qT,
                                                  const _Float16* __restrict__ kT,
                                                  const _Float16* __restrict__ vbuf,
                                                  _Float16* __restrict__ aoT) {
  __shared__ __attribute__((aligned(32))) float smem[16 * 512];   // 32 KB
  float*    dots  = smem;
  _Float16* attnh = (_Float16*)smem;

  int lane = threadIdx.x;
  int blk  = blockIdx.x;          // 4 * 8 * 256
  int b  = blk >> 11;
  int h  = (blk >> 8) & 7;
  int it = blk & 255;

  const _Float16* qbase = qT   + ((size_t)(b * 8 + h) * 4096) * 64;
  const _Float16* kbase = kT   + ((size_t)(b * 8 + h) * 512) * 64;
  const _Float16* vbase = vbuf + ((size_t)(b * 8 + h) * 64) * 512;

  const int l15 = lane & 15;
  const int rb  = (lane >> 4) << 3;
  const int klo = (lane >> 4) << 4;
  const int kb  = (lane >> 4) << 4;

  // ---- Phase 1: dots^T = K x q_tile^T; q fragments hoisted ----
  const _Float16* qrow = qbase + (size_t)(it * 16 + l15) * 64;
  v16h b0 = *(const v16h*)(qrow + kb);
  v16h b1 = *(const v16h*)(qrow + 32 + kb);
  for (int jt = 0; jt < 32; ++jt) {
    const _Float16* at = kbase + (size_t)(jt * 16 + l15) * 64 + klo;
    v16h a0 = cat8(*(const v8h*)(at),      *(const v8h*)(at + 8));
    v16h a1 = cat8(*(const v8h*)(at + 32), *(const v8h*)(at + 40));
    v8f acc = {};
    acc = wmma_step(a0, b0, acc);
    acc = wmma_step(a1, b1, acc);
    float* dp = dots + (size_t)l15 * 512 + jt * 16 + rb;   // 8 contiguous f32
    #pragma unroll
    for (int r = 0; r < 8; ++r) dp[r] = acc[r] * 0.125f;   // 1/sqrt(64)
  }
  __syncthreads();

  // ---- Phase 2: softmax over 512, all lanes (row l15, half lane>>4);
  //      f16 attn packed in place over the dots buffer ----
  {
    float* rowp = dots + (size_t)l15 * 512 + ((lane >> 4) << 8);
    float mx = -3.0e38f;
    for (int j = 0; j < 256; ++j) mx = fmaxf(mx, rowp[j]);
    mx = fmaxf(mx, __shfl_xor(mx, 16, 32));
    float sum = 0.f;
    for (int j = 0; j < 256; ++j) { float e = __expf(rowp[j] - mx); sum += e; rowp[j] = e; }
    sum += __shfl_xor(sum, 16, 32);
    float inv = 1.f / sum;
    // forward in-place pack: write half j/2 after reading float j (safe)
    _Float16* a16 = attnh + (size_t)l15 * 1024 + ((lane >> 4) << 9);
    for (int j = 0; j < 256; ++j) a16[j] = (_Float16)(rowp[j] * inv);
  }
  __syncthreads();

  // ---- Phase 3: out^T = V x attn^T; 4 d-tiles share the attn fragment ----
  v8f acc[4] = {};
  const _Float16* brow = attnh + (size_t)l15 * 1024;        // chunked attn row
  const _Float16* ap0  = vbase + (size_t)l15 * 512 + klo;   // global A row
  for (int k = 0; k < 512; k += 32) {
    int kk = k + kb;
    v16h bf = *(const v16h*)(brow + ((kk >> 8) << 9) + (kk & 255)); // ds_load_b128 x2
    #pragma unroll
    for (int mt = 0; mt < 4; ++mt) {
      const _Float16* at = ap0 + (size_t)mt * 16 * 512;
      v16h a = cat8(*(const v8h*)(at + k), *(const v8h*)(at + k + 8));
      acc[mt] = wmma_step(a, bf, acc[mt]);
    }
  }
  int i = it * 16 + l15;
  #pragma unroll
  for (int mt = 0; mt < 4; ++mt) {
    _Float16* dst = aoT + ((size_t)b * 4096 + i) * 512 + h * 64 + mt * 16 + rb;
    #pragma unroll
    for (int r = 0; r < 8; ++r) dst[r] = (_Float16)acc[mt][r];  // b128 store
  }
}

// ---------------------------------------------------------------------------
// Output pointwise conv: y = Wout(128x512) x ao[b](512x4096) + b_out
// 4 M-tiles per wave.  grid: 256 x 256 (2048 wave tasks = 4 x 2 x 256)
// ---------------------------------------------------------------------------
__global__ __launch_bounds__(256) void gemm_out(const _Float16* __restrict__ Wout,
                                                const _Float16* __restrict__ aoT,
                                                const float* __restrict__ b_out,
                                                float* __restrict__ y) {
  int lane = threadIdx.x & 31;
  int task = blockIdx.x * 8 + (threadIdx.x >> 5);
  int b  = task >> 9;
  int t  = task & 511;
  int mq = t >> 8;             // 0..1
  int nt = t & 255;
  v8f acc[4] = {};
  gemm_tile4(Wout, 512, aoT + (size_t)b * 4096 * 512, 512,
             mq * 64 + (lane & 15), nt * 16 + (lane & 15), 512, lane, acc);
  int s  = nt * 16 + (lane & 15);
  int rb = (lane >> 4) << 3;
  #pragma unroll
  for (int tt = 0; tt < 4; ++tt) {
    #pragma unroll
    for (int r = 0; r < 8; ++r) {
      int o = mq * 64 + tt * 16 + rb + r;
      y[((size_t)b * 128 + o) * 4096 + s] = acc[tt][r] + b_out[o];
    }
  }
}

// ---------------------------------------------------------------------------
// Host launch
// ---------------------------------------------------------------------------
extern "C" void kernel_launch(void* const* d_in, const int* in_sizes, int n_in,
                              void* d_out, int out_size, void* d_ws, size_t ws_size,
                              hipStream_t stream) {
  const float* x       = (const float*)d_in[0];
  const float* wq_dw   = (const float*)d_in[1];
  const float* bn_q_g  = (const float*)d_in[2];
  const float* bn_q_b  = (const float*)d_in[3];
  const float* bn_q_m  = (const float*)d_in[4];
  const float* bn_q_v  = (const float*)d_in[5];
  const float* wq_pw   = (const float*)d_in[6];
  const float* wkv_dw  = (const float*)d_in[7];
  const float* bn_kv_g = (const float*)d_in[8];
  const float* bn_kv_b = (const float*)d_in[9];
  const float* bn_kv_m = (const float*)d_in[10];
  const float* bn_kv_v = (const float*)d_in[11];
  const float* wkv_pw  = (const float*)d_in[12];
  const float* w_out   = (const float*)d_in[13];
  const float* b_out   = (const float*)d_in[14];
  float* y = (float*)d_out;

  char* ws = (char*)d_ws;
  size_t off = 0;
  _Float16* Wq16   = (_Float16*)(ws + off); off += (size_t)512 * 128 * 2;
  _Float16* Wkv16  = (_Float16*)(ws + off); off += (size_t)1024 * 128 * 2;
  _Float16* Wout16 = (_Float16*)(ws + off); off += (size_t)128 * 512 * 2;
  _Float16* qdwT   = (_Float16*)(ws + off); off += (size_t)4 * 4096 * 128 * 2;
  _Float16* kvdwT  = (_Float16*)(ws + off); off += (size_t)4 * 512 * 128 * 2;
  _Float16* qT     = (_Float16*)(ws + off); off += (size_t)4 * 8 * 4096 * 64 * 2;
  _Float16* kT     = (_Float16*)(ws + off); off += (size_t)4 * 8 * 512 * 64 * 2;
  _Float16* vbuf   = (_Float16*)(ws + off); off += (size_t)4 * 8 * 64 * 512 * 2;
  _Float16* aoT    = (_Float16*)(ws + off); off += (size_t)4 * 4096 * 512 * 2;

  // Weight conversions to f16
  cvt_f32_f16<<<(512 * 128 + 255) / 256, 256, 0, stream>>>(wq_pw, Wq16, 512 * 128);
  cvt_f32_f16<<<(1024 * 128 + 255) / 256, 256, 0, stream>>>(wkv_pw, Wkv16, 1024 * 128);
  cvt_f32_f16<<<(128 * 512 + 255) / 256, 256, 0, stream>>>(w_out, Wout16, 128 * 512);

  // Depthwise convs + BN (transposed f16 outputs)
  dwconv_bn<16, 1><<<(4 * 128 * 4096 + 255) / 256, 256, 0, stream>>>(
      x, wq_dw, bn_q_g, bn_q_b, bn_q_m, bn_q_v, qdwT);
  dwconv_bn<8, 2><<<(4 * 128 * 512 + 255) / 256, 256, 0, stream>>>(
      x, wkv_dw, bn_kv_g, bn_kv_b, bn_kv_m, bn_kv_v, kvdwT);

  // Pointwise conv GEMMs (WMMA, 4 M-tiles per wave)
  gemm_q<<<1024, 256, 0, stream>>>(Wq16, qdwT, qT);
  gemm_kv<<<256, 256, 0, stream>>>(Wkv16, kvdwT, kT, vbuf);

  // Fused attention (WMMA + in-LDS softmax, 32 KB LDS)
  attn_kernel<<<4 * 8 * 256, 32, 0, stream>>>(qT, kT, vbuf, aoT);

  // Output conv GEMM (WMMA) + bias
  gemm_out<<<256, 256, 0, stream>>>(Wout16, aoT, b_out, y);

  (void)in_sizes; (void)n_in; (void)out_size; (void)ws_size;
}
